// VQVAE_GT_10402410790980
// MI455X (gfx1250) — matrix operand order repaired and checked by reading, not compiled, not run on hardware
//
#include <hip/hip_runtime.h>
#include <hip/hip_bf16.h>

typedef __bf16 bf16;
typedef __attribute__((ext_vector_type(16))) __bf16 bf16x16;
typedef __attribute__((ext_vector_type(8)))  float   f32x8;
typedef __attribute__((ext_vector_type(4)))  float   f32x4;

union V32 { bf16x16 v; f32x4 f[2]; };

#define TOK    512     // B*N tokens
#define NPOS   128     // N
#define DIMC   256
#define INNERC 512
#define EDGEC  64
#define HEADSC 8
#define DHC    64
#define KCODES 512

// ---------------------------------------------------------------------------
// Edge LayerNorm over channel dim: edges (B,64,128,128) f32 -> e_bf (B,128,128,64) bf16
// ---------------------------------------------------------------------------
__global__ __launch_bounds__(256)
void k_edge_ln(const float* __restrict__ edges, const float* __restrict__ g,
               const float* __restrict__ be, bf16* __restrict__ e_bf)
{
  const int t  = blockIdx.x * 256 + threadIdx.x;   // flat (b,i,j), 65536 total
  const int b  = t >> 14;
  const int ij = t & 16383;
  const float* base = edges + (size_t)b * EDGEC * 16384 + ij;
  float vals[EDGEC];
  float m = 0.f;
  #pragma unroll
  for (int c = 0; c < EDGEC; ++c) { vals[c] = base[(size_t)c * 16384]; m += vals[c]; }
  m *= (1.f / 64.f);
  float v = 0.f;
  #pragma unroll
  for (int c = 0; c < EDGEC; ++c) { float d = vals[c] - m; v += d * d; }
  v *= (1.f / 64.f);
  const float r = rsqrtf(v + 1e-5f);
  bf16* dst = e_bf + (size_t)t * EDGEC;
  #pragma unroll
  for (int c = 0; c < EDGEC; ++c)
    dst[c] = (bf16)((vals[c] - m) * r * g[c] + be[c]);
}

// ---------------------------------------------------------------------------
// Weight transpose+convert: W (K x N) f32 -> WT (N x K) bf16
// ---------------------------------------------------------------------------
__global__ __launch_bounds__(256)
void k_wt(const float* __restrict__ W, bf16* __restrict__ WT, int K, int Nc)
{
  const int idx = blockIdx.x * 256 + threadIdx.x;
  if (idx >= K * Nc) return;
  const int n = idx / K, k = idx % K;
  WT[idx] = (bf16)W[(size_t)k * Nc + n];
}

// f32 -> bf16 convert
__global__ __launch_bounds__(256)
void k_f2b(const float* __restrict__ x, bf16* __restrict__ y, int cnt)
{
  const int i = blockIdx.x * 256 + threadIdx.x;
  if (i < cnt) y[i] = (bf16)x[i];
}

// GELU(exact) f32 -> bf16
__global__ __launch_bounds__(256)
void k_gelu(const float* __restrict__ x, bf16* __restrict__ y, int cnt)
{
  const int i = blockIdx.x * 256 + threadIdx.x;
  if (i < cnt) {
    float v = x[i];
    y[i] = (bf16)(0.5f * v * (1.f + erff(v * 0.70710678118654752f)));
  }
}

// ---------------------------------------------------------------------------
// LayerNorm over 256 dims: x (T x 256) f32 -> y bf16 (GEMM-ready)
// ---------------------------------------------------------------------------
__global__ __launch_bounds__(256)
void k_ln256(const float* __restrict__ x, const float* __restrict__ g,
             const float* __restrict__ be, bf16* __restrict__ y)
{
  __shared__ float red[256];
  const int t = blockIdx.x, d = threadIdx.x;
  const float v = x[(size_t)t * DIMC + d];
  red[d] = v; __syncthreads();
  for (int s = 128; s > 0; s >>= 1) { if (d < s) red[d] += red[d + s]; __syncthreads(); }
  const float m = red[0] * (1.f / 256.f);
  __syncthreads();
  const float df = v - m;
  red[d] = df * df; __syncthreads();
  for (int s = 128; s > 0; s >>= 1) { if (d < s) red[d] += red[d + s]; __syncthreads(); }
  const float var = red[0] * (1.f / 256.f);
  y[(size_t)t * DIMC + d] = (bf16)(df * rsqrtf(var + 1e-5f) * g[d] + be[d]);
}

// ---------------------------------------------------------------------------
// BF16 WMMA GEMM: C(MxN) = A(MxK) @ BT(NxK)^T + bias
// 8 waves/block; wave computes a 16x64 slab (4 acc tiles, A reused 4x).
// Block tile = 64(M) x 128(N). Requires M%64==0, N%128==0, K%32==0.
// ---------------------------------------------------------------------------
__global__ __launch_bounds__(256)
void k_gemm(const bf16* __restrict__ A, const bf16* __restrict__ BT,
            const float* __restrict__ bias, float* __restrict__ C,
            int Nc, int K)
{
  const int tid = threadIdx.x;
  const int wv = tid >> 5, lane = tid & 31;
  const int half = (lane >> 4) & 1, l = lane & 15;
  const int tm  = blockIdx.x * 64  + (wv & 3) * 16;
  const int tnb = blockIdx.y * 128 + (wv >> 2) * 64;
  const bf16* arow = A + (size_t)(tm + l) * K;
  const bf16* b0r  = BT + (size_t)(tnb      + l) * K;
  const bf16* b1r  = BT + (size_t)(tnb + 16 + l) * K;
  const bf16* b2r  = BT + (size_t)(tnb + 32 + l) * K;
  const bf16* b3r  = BT + (size_t)(tnb + 48 + l) * K;
  f32x8 acc0 = {}, acc1 = {}, acc2 = {}, acc3 = {};
  for (int kk = 0; kk < K; kk += 32) {
    V32 ua;
    ua.f[0] = *(const f32x4*)(arow + kk + half * 16);
    ua.f[1] = *(const f32x4*)(arow + kk + half * 16 + 8);
    V32 ub0, ub1, ub2, ub3;
    ub0.f[0] = *(const f32x4*)(b0r + kk + half * 16);
    ub0.f[1] = *(const f32x4*)(b0r + kk + half * 16 + 8);
    ub1.f[0] = *(const f32x4*)(b1r + kk + half * 16);
    ub1.f[1] = *(const f32x4*)(b1r + kk + half * 16 + 8);
    ub2.f[0] = *(const f32x4*)(b2r + kk + half * 16);
    ub2.f[1] = *(const f32x4*)(b2r + kk + half * 16 + 8);
    ub3.f[0] = *(const f32x4*)(b3r + kk + half * 16);
    ub3.f[1] = *(const f32x4*)(b3r + kk + half * 16 + 8);
    acc0 = __builtin_amdgcn_wmma_f32_16x16x32_bf16(false, ua.v, false, ub0.v, (short)0, acc0, false, false);
    acc1 = __builtin_amdgcn_wmma_f32_16x16x32_bf16(false, ua.v, false, ub1.v, (short)0, acc1, false, false);
    acc2 = __builtin_amdgcn_wmma_f32_16x16x32_bf16(false, ua.v, false, ub2.v, (short)0, acc2, false, false);
    acc3 = __builtin_amdgcn_wmma_f32_16x16x32_bf16(false, ua.v, false, ub3.v, (short)0, acc3, false, false);
  }
  float* crow = C + (size_t)(tm + half * 8) * Nc + tnb + l;
  const float bb0 = bias[tnb + l],      bb1 = bias[tnb + 16 + l];
  const float bb2 = bias[tnb + 32 + l], bb3 = bias[tnb + 48 + l];
  #pragma unroll
  for (int r = 0; r < 8; ++r) {
    crow[(size_t)r * Nc]      = acc0[r] + bb0;
    crow[(size_t)r * Nc + 16] = acc1[r] + bb1;
    crow[(size_t)r * Nc + 32] = acc2[r] + bb2;
    crow[(size_t)r * Nc + 48] = acc3[r] + bb3;
  }
}

// ---------------------------------------------------------------------------
// Rotary (interleaved pairs) applied in place to q and to k-half of kv
// ---------------------------------------------------------------------------
__global__ __launch_bounds__(256)
void k_rotary(float* __restrict__ q, float* __restrict__ kv)
{
  const int t = blockIdx.x, n = t & 127;
  const int p = threadIdx.x;          // pair 0..255
  const int o = p * 2, d = o & 63;    // dim-in-head of pair start (even)
  const float fr = (float)n * powf(10000.f, -(float)d * (1.f / 64.f));
  const float c = cosf(fr), s = sinf(fr);
  float* qr = q  + (size_t)t * INNERC + o;
  float* kr = kv + (size_t)t * 1024   + o;
  float x0 = qr[0], x1 = qr[1];
  qr[0] = x0 * c - x1 * s; qr[1] = x1 * c + x0 * s;
  x0 = kr[0]; x1 = kr[1];
  kr[0] = x0 * c - x1 * s; kr[1] = x1 * c + x0 * s;
}

// ---------------------------------------------------------------------------
// Fused edge-conditioned attention. One block per (b,i). Per head:
//   WMMA: ekv (128j x 64d) = e_ln[b,i] (128x64) @ w_ekv_h^T  (+b_ekv)  -> LDS
//   (A fragments are h/dt-invariant: hoisted into registers once per block)
//   VALU: sim_j = q . (k_j + ekv_j); softmax; out = sum att (v_j + ekv_j)
// ---------------------------------------------------------------------------
__global__ __launch_bounds__(256)
void k_attn(const bf16* __restrict__ e_bf, const bf16* __restrict__ wekvT,
            const float* __restrict__ b_ekv, const float* __restrict__ q,
            const float* __restrict__ kv, float* __restrict__ attn_out)
{
  __shared__ float ekv_s[NPOS * DHC];   // 32 KB
  __shared__ float qs[DHC];
  __shared__ float sim_s[NPOS];
  __shared__ float att_s[NPOS];
  __shared__ float part_s[4 * DHC];
  __shared__ float smax_s, sinv_s;

  const int t = blockIdx.x;             // b*128 + i
  const int b = t >> 7;
  const int tid = threadIdx.x;
  const int wv = tid >> 5, lane = tid & 31;
  const int half = (lane >> 4) & 1, l = lane & 15;

  const bf16* e_row = e_bf + (size_t)t * NPOS * EDGEC;
  const bf16* arow  = e_row + (size_t)(wv * 16 + l) * EDGEC;

  // hoist A fragments (K=64 -> two k-steps), invariant across heads/d-tiles
  V32 ua0, ua1;
  ua0.f[0] = *(const f32x4*)(arow + half * 16);
  ua0.f[1] = *(const f32x4*)(arow + half * 16 + 8);
  ua1.f[0] = *(const f32x4*)(arow + 32 + half * 16);
  ua1.f[1] = *(const f32x4*)(arow + 32 + half * 16 + 8);

  for (int h = 0; h < HEADSC; ++h) {
    // ---- WMMA: wave wv owns j-rows [wv*16, wv*16+16), loops 4 d-tiles ----
    for (int dt = 0; dt < 4; ++dt) {
      const bf16* bcol = wekvT + (size_t)(h * DHC + dt * 16 + l) * EDGEC;
      V32 ub0, ub1;
      ub0.f[0] = *(const f32x4*)(bcol + half * 16);
      ub0.f[1] = *(const f32x4*)(bcol + half * 16 + 8);
      ub1.f[0] = *(const f32x4*)(bcol + 32 + half * 16);
      ub1.f[1] = *(const f32x4*)(bcol + 32 + half * 16 + 8);
      f32x8 acc = {};
      acc = __builtin_amdgcn_wmma_f32_16x16x32_bf16(false, ua0.v, false, ub0.v, (short)0, acc, false, false);
      acc = __builtin_amdgcn_wmma_f32_16x16x32_bf16(false, ua1.v, false, ub1.v, (short)0, acc, false, false);
      const float bb = b_ekv[h * DHC + dt * 16 + l];
      #pragma unroll
      for (int r = 0; r < 8; ++r)
        ekv_s[(wv * 16 + half * 8 + r) * DHC + dt * 16 + l] = acc[r] + bb;
    }
    if (tid < DHC) qs[tid] = q[(size_t)t * INNERC + h * DHC + tid];
    __syncthreads();

    // ---- sim: thread pair per j, each half sums 32 dims ----
    {
      const int j = tid >> 1, p = tid & 1;
      const float* krow = kv + (size_t)((b << 7) + j) * 1024 + h * DHC;
      const float* er   = ekv_s + j * DHC;
      float s = 0.f;
      for (int d = p * 32; d < p * 32 + 32; ++d)
        s += qs[d] * (krow[d] + er[d]);
      s += __shfl_xor(s, 1);
      if (p == 0) sim_s[j] = s * 0.125f;   // DH^-0.5
    }
    __syncthreads();
    if (tid < 32) {
      float m = fmaxf(fmaxf(sim_s[tid], sim_s[tid + 32]),
                      fmaxf(sim_s[tid + 64], sim_s[tid + 96]));
      for (int off = 16; off > 0; off >>= 1) m = fmaxf(m, __shfl_xor(m, off));
      if (tid == 0) smax_s = m;
    }
    __syncthreads();
    if (tid < NPOS) att_s[tid] = expf(sim_s[tid] - smax_s);
    __syncthreads();
    if (tid < 32) {
      float s2 = att_s[tid] + att_s[tid + 32] + att_s[tid + 64] + att_s[tid + 96];
      for (int off = 16; off > 0; off >>= 1) s2 += __shfl_xor(s2, off);
      if (tid == 0) sinv_s = 1.f / s2;
    }
    __syncthreads();

    // ---- out: thread (d, j-quarter) ----
    {
      const int d = tid & 63, part = tid >> 6;
      float o = 0.f;
      for (int j = part * 32; j < part * 32 + 32; ++j) {
        const float* vrow = kv + (size_t)((b << 7) + j) * 1024 + INNERC + h * DHC;
        o += att_s[j] * (vrow[d] + ekv_s[j * DHC + d]);
      }
      part_s[part * DHC + d] = o;
    }
    __syncthreads();
    if (tid < DHC)
      attn_out[(size_t)t * INNERC + h * DHC + tid] =
        (part_s[tid] + part_s[DHC + tid] + part_s[2 * DHC + tid] + part_s[3 * DHC + tid]) * sinv_s;
    __syncthreads();    // protect LDS reuse next head
  }
}

// ---------------------------------------------------------------------------
// Gated residual (token-scalar gate): res = x*g + res*(1-g), g = sigmoid(.)
// ---------------------------------------------------------------------------
__global__ __launch_bounds__(256)
void k_gated(const float* __restrict__ x, float* __restrict__ res,
             const float* __restrict__ gw)
{
  __shared__ float red[256];
  const int t = blockIdx.x, d = threadIdx.x;
  const float xv = x[(size_t)t * DIMC + d];
  const float rv = res[(size_t)t * DIMC + d];
  red[d] = xv * gw[d] + rv * gw[DIMC + d] + (xv - rv) * gw[2 * DIMC + d];
  __syncthreads();
  for (int s = 128; s > 0; s >>= 1) { if (d < s) red[d] += red[d + s]; __syncthreads(); }
  const float gt = 1.f / (1.f + expf(-red[0]));
  res[(size_t)t * DIMC + d] = xv * gt + rv * (1.f - gt);
}

// ---------------------------------------------------------------------------
// VQ bias: -0.5 * ||c_k||^2  (so score = z.c - 0.5||c||^2; argmax == argmin dist)
// ---------------------------------------------------------------------------
__global__ __launch_bounds__(256)
void k_c2(const float* __restrict__ cb, float* __restrict__ c2)
{
  const int k = blockIdx.x * 256 + threadIdx.x;
  if (k >= KCODES) return;
  const float* row = cb + (size_t)k * DIMC;
  float s = 0.f;
  for (int d = 0; d < DIMC; ++d) s += row[d] * row[d];
  c2[k] = -0.5f * s;
}

// ---------------------------------------------------------------------------
// VQ pick: argmax score row (ties -> lowest index), gather codebook row to out
// ---------------------------------------------------------------------------
__global__ __launch_bounds__(256)
void k_vq_pick(const float* __restrict__ scores, const float* __restrict__ cb,
               float* __restrict__ out)
{
  __shared__ float bd[256];
  __shared__ int   bi[256];
  const int t = blockIdx.x, tid = threadIdx.x;
  const float* row = scores + (size_t)t * KCODES;
  float best = row[tid]; int bidx = tid;
  { const float s2 = row[tid + 256]; if (s2 > best) { best = s2; bidx = tid + 256; } }
  bd[tid] = best; bi[tid] = bidx;
  __syncthreads();
  for (int s = 128; s > 0; s >>= 1) {
    if (tid < s) {
      if (bd[tid + s] > bd[tid] || (bd[tid + s] == bd[tid] && bi[tid + s] < bi[tid])) {
        bd[tid] = bd[tid + s]; bi[tid] = bi[tid + s];
      }
    }
    __syncthreads();
  }
  const int idx = bi[0];
  out[(size_t)t * DIMC + tid] = cb[(size_t)idx * DIMC + tid];
}

// ---------------------------------------------------------------------------
extern "C" void kernel_launch(void* const* d_in, const int* in_sizes, int n_in,
                              void* d_out, int out_size, void* d_ws, size_t ws_size,
                              hipStream_t stream)
{
  (void)in_sizes; (void)n_in; (void)out_size; (void)ws_size;
  const float* in_nodes = (const float*)d_in[0];
  const float* edges    = (const float*)d_in[1];
  const float* eg    = (const float*)d_in[2];
  const float* eb    = (const float*)d_in[3];
  const float* ln1g  = (const float*)d_in[4];
  const float* ln1b  = (const float*)d_in[5];
  const float* w_exp = (const float*)d_in[6];
  const float* b_exp = (const float*)d_in[7];
  const float* w_q   = (const float*)d_in[8];
  const float* b_q   = (const float*)d_in[9];
  const float* w_kv  = (const float*)d_in[10];
  const float* b_kv  = (const float*)d_in[11];
  const float* w_ekv = (const float*)d_in[12];
  const float* b_ekv = (const float*)d_in[13];
  const float* w_out = (const float*)d_in[14];
  const float* b_out = (const float*)d_in[15];
  const float* g1    = (const float*)d_in[16];
  const float* ln2g  = (const float*)d_in[17];
  const float* ln2b  = (const float*)d_in[18];
  const float* w_ff1 = (const float*)d_in[19];
  const float* b_ff1 = (const float*)d_in[20];
  const float* w_ff2 = (const float*)d_in[21];
  const float* b_ff2 = (const float*)d_in[22];
  const float* g2    = (const float*)d_in[23];
  const float* cb    = (const float*)d_in[24];

  // ---- workspace carve-out ----
  char* p = (char*)d_ws;
  auto alloc = [&](size_t bytes) -> void* {
    void* r = (void*)p; p += (bytes + 255) & ~(size_t)255; return r;
  };
  float* nodes    = (float*)alloc((size_t)TOK * DIMC * 4);
  bf16*  e_bf     = (bf16*) alloc((size_t)TOK * NPOS * EDGEC * 2);   // 8 MB
  bf16*  xln_bf   = (bf16*) alloc((size_t)TOK * DIMC * 2);
  float* x_f      = (float*)alloc((size_t)TOK * INNERC * 4);
  bf16*  x_bf     = (bf16*) alloc((size_t)TOK * INNERC * 2);
  float* q_f      = (float*)alloc((size_t)TOK * INNERC * 4);
  float* kv_f     = (float*)alloc((size_t)TOK * 1024 * 4);
  float* attn_f   = (float*)alloc((size_t)TOK * INNERC * 4);
  bf16*  attn_bf  = (bf16*) alloc((size_t)TOK * INNERC * 2);
  float* proj_f   = (float*)alloc((size_t)TOK * DIMC * 4);
  bf16*  hln_bf   = (bf16*) alloc((size_t)TOK * DIMC * 2);
  float* ff1_f    = (float*)alloc((size_t)TOK * 1024 * 4);
  bf16*  mid_bf   = (bf16*) alloc((size_t)TOK * 1024 * 2);
  float* c2_f     = (float*)alloc((size_t)KCODES * 4);
  float* score_f  = (float*)alloc((size_t)TOK * KCODES * 4);         // 1 MB
  bf16*  z_bf     = (bf16*) alloc((size_t)TOK * DIMC * 2);
  bf16*  cb_bf    = (bf16*) alloc((size_t)KCODES * DIMC * 2);
  bf16*  w_expT   = (bf16*) alloc((size_t)INNERC * DIMC * 2);
  bf16*  w_qT     = (bf16*) alloc((size_t)INNERC * INNERC * 2);
  bf16*  w_kvT    = (bf16*) alloc((size_t)1024 * INNERC * 2);
  bf16*  w_ekvT   = (bf16*) alloc((size_t)INNERC * EDGEC * 2);
  bf16*  w_outT   = (bf16*) alloc((size_t)DIMC * INNERC * 2);
  bf16*  w_ff1T   = (bf16*) alloc((size_t)1024 * DIMC * 2);
  bf16*  w_ff2T   = (bf16*) alloc((size_t)DIMC * 1024 * 2);

  // nodes working copy (don't mutate inputs)
  hipMemcpyAsync(nodes, in_nodes, (size_t)TOK * DIMC * 4, hipMemcpyDeviceToDevice, stream);

  // one-time prep: edge LN, VQ bias, codebook bf16 (already N x K layout)
  k_edge_ln<<<65536 / 256, 256, 0, stream>>>(edges, eg, eb, e_bf);
  k_c2<<<2, 256, 0, stream>>>(cb, c2_f);
  k_f2b<<<(KCODES * DIMC + 255) / 256, 256, 0, stream>>>(cb, cb_bf, KCODES * DIMC);

  for (int i = 0; i < 2; ++i) {
    // transpose+convert this layer's weights to bf16 (N x K)
    k_wt<<<(DIMC * INNERC + 255) / 256, 256, 0, stream>>>(w_exp + (size_t)i * DIMC * INNERC, w_expT, DIMC, INNERC);
    k_wt<<<(INNERC * INNERC + 255) / 256, 256, 0, stream>>>(w_q + (size_t)i * INNERC * INNERC, w_qT, INNERC, INNERC);
    k_wt<<<(INNERC * 1024 + 255) / 256, 256, 0, stream>>>(w_kv + (size_t)i * INNERC * 1024, w_kvT, INNERC, 1024);
    k_wt<<<(EDGEC * INNERC + 255) / 256, 256, 0, stream>>>(w_ekv + (size_t)i * EDGEC * INNERC, w_ekvT, EDGEC, INNERC);
    k_wt<<<(INNERC * DIMC + 255) / 256, 256, 0, stream>>>(w_out + (size_t)i * INNERC * DIMC, w_outT, INNERC, DIMC);
    k_wt<<<(DIMC * 1024 + 255) / 256, 256, 0, stream>>>(w_ff1 + (size_t)i * DIMC * 1024, w_ff1T, DIMC, 1024);
    k_wt<<<(1024 * DIMC + 255) / 256, 256, 0, stream>>>(w_ff2 + (size_t)i * 1024 * DIMC, w_ff2T, 1024, DIMC);

    // PreNorm + projections (WMMA GEMMs, block tile 64x128)
    k_ln256<<<TOK, 256, 0, stream>>>(nodes, ln1g + i * DIMC, ln1b + i * DIMC, xln_bf);
    k_gemm<<<dim3(TOK / 64, INNERC / 128), 256, 0, stream>>>(xln_bf, w_expT, b_exp + i * INNERC, x_f, INNERC, DIMC);
    k_f2b<<<(TOK * INNERC + 255) / 256, 256, 0, stream>>>(x_f, x_bf, TOK * INNERC);
    k_gemm<<<dim3(TOK / 64, INNERC / 128), 256, 0, stream>>>(x_bf, w_qT, b_q + i * INNERC, q_f, INNERC, INNERC);
    k_gemm<<<dim3(TOK / 64, 1024 / 128), 256, 0, stream>>>(x_bf, w_kvT, b_kv + i * 1024, kv_f, 1024, INNERC);
    k_rotary<<<TOK, 256, 0, stream>>>(q_f, kv_f);

    // fused edge-conditioned attention (WMMA ekv + VALU softmax)
    k_attn<<<TOK, 256, 0, stream>>>(e_bf, w_ekvT, b_ekv + i * INNERC, q_f, kv_f, attn_f);

    // out projection + gated residual
    k_f2b<<<(TOK * INNERC + 255) / 256, 256, 0, stream>>>(attn_f, attn_bf, TOK * INNERC);
    k_gemm<<<dim3(TOK / 64, DIMC / 128), 256, 0, stream>>>(attn_bf, w_outT, b_out + i * DIMC, proj_f, DIMC, INNERC);
    k_gated<<<TOK, 256, 0, stream>>>(proj_f, nodes, g1 + i * 3 * DIMC);

    // PreNorm + FeedForward + gated residual
    k_ln256<<<TOK, 256, 0, stream>>>(nodes, ln2g + i * DIMC, ln2b + i * DIMC, hln_bf);
    k_gemm<<<dim3(TOK / 64, 1024 / 128), 256, 0, stream>>>(hln_bf, w_ff1T, b_ff1 + i * 1024, ff1_f, 1024, DIMC);
    k_gelu<<<(TOK * 1024 + 255) / 256, 256, 0, stream>>>(ff1_f, mid_bf, TOK * 1024);
    k_gemm<<<dim3(TOK / 64, DIMC / 128), 256, 0, stream>>>(mid_bf, w_ff2T, b_ff2 + i * DIMC, proj_f, DIMC, 1024);
    k_gated<<<TOK, 256, 0, stream>>>(proj_f, nodes, g2 + i * 3 * DIMC);
  }

  // VQ as WMMA GEMM: score = z @ cb^T - 0.5||c||^2, then argmax + gather
  k_f2b<<<(TOK * DIMC + 255) / 256, 256, 0, stream>>>(nodes, z_bf, TOK * DIMC);
  k_gemm<<<dim3(TOK / 64, KCODES / 128), 256, 0, stream>>>(z_bf, cb_bf, c2_f, score_f, KCODES, DIMC);
  k_vq_pick<<<TOK, 256, 0, stream>>>(score_f, cb, (float*)d_out);
}
